// QuantizeUnsignedActPercentage_66838281060427
// MI455X (gfx1250) — compile-verified, matrix operand order checked by reading, not run on hardware
//
#include <hip/hip_runtime.h>
#include <cstdint>
#include <cstddef>

// ---- problem constants (x: [16,128,112,112] f32) ----
#define C_CH     128
#define SLICES   2048        // 16*128 contiguous (n,c) slices
#define SLICE_F  12544       // 112*112 floats per slice
#define SLICE_V4 3136        // SLICE_F / 4
#define FULL_CHUNKS 12       // 12 * 1024 = 12288 floats; tail = 256 floats
#define TAIL_BASE   12288
#define CH_SIZE  200704.0f   // 16*112*112 elements per channel
#define P_LO_F   0.01f
#define P_HI_F   0.02f
#define Q_MAX_F  255.0f

typedef float f32x4 __attribute__((ext_vector_type(4)));
typedef int   v8i   __attribute__((ext_vector_type(8)));

// ---------------- workspace init (ws poisoned with 0xAA each run) ------------
__global__ void k_init(unsigned* flag, unsigned* cnt, unsigned* gmax) {
  const int t = threadIdx.x;
  if (t == 0) *flag = 0u;
  if (t < C_CH) { cnt[t] = 0u; gmax[t] = 0u; }  // x >= 0 -> uint-ordered max, init 0
}

// ---------------- pass 1: per-channel max (DRAM-bound pass) ------------------
__global__ void k_max(const float* __restrict__ x, unsigned* __restrict__ gmax) {
  const int s = blockIdx.x;            // slice = (n,c)
  const int c = s & (C_CH - 1);
  const f32x4* __restrict__ src = (const f32x4*)(x + (size_t)s * SLICE_F);
  float m = 0.0f;
  for (int i = threadIdx.x; i < SLICE_V4; i += 256) {
    __builtin_prefetch((const void*)(src + i + 512), 0, 1); // global_prefetch_b8
    f32x4 v = src[i];
    m = fmaxf(m, fmaxf(fmaxf(v.x, v.y), fmaxf(v.z, v.w)));
  }
  __shared__ unsigned sm;
  if (threadIdx.x == 0) sm = 0u;
  __syncthreads();
  atomicMax(&sm, __float_as_uint(m));  // non-negative floats: uint order == float order
  __syncthreads();
  if (threadIdx.x == 0) atomicMax(&gmax[c], sm);
}

// ---------------- seed bisection state ---------------------------------------
__global__ void k_seed(const unsigned* gmax, float* l, float* r,
                       unsigned* cnt, unsigned* flag) {
  const int c = threadIdx.x;           // 128 threads
  l[c] = 0.0f;
  r[c] = __uint_as_float(gmax[c]);
  cnt[c] = 0u;
  if (c == 0) *flag = 0u;
}

__device__ __forceinline__ int pack4(f32x4 f, float p) {
  return (f.x > p ? 0x00000001 : 0) | (f.y > p ? 0x00000100 : 0) |
         (f.z > p ? 0x00010000 : 0) | (f.w > p ? 0x01000000 : 0);
}

// ---------------- counting pass: #elements > pivot, via WMMA IU8 -------------
// Each wave packs 1024 comparison results (1 byte each) into an A matrix
// (16x64 IU8, 8 VGPRs/lane) and multiplies by an all-ones B: every column of
// D holds the A row sums, so one v_wmma_i32_16x16x64_iu8 reduces 1024 bools.
// Remainder (256 floats) is peeled so the hot loop has NO per-load guards:
// 8 unconditional global_load_b128 issue back-to-back (full MLP), and all
// branches are wave-uniform (EXEC all-ones for WMMA).
__global__ void k_count(const float* __restrict__ x,
                        const float* __restrict__ l, const float* __restrict__ r,
                        const unsigned* __restrict__ flag, unsigned* __restrict__ cnt) {
  if (*flag) return;                   // converged: whole pass is a no-op
  const int s = blockIdx.x;
  const int c = s & (C_CH - 1);
  const float pivot = 0.5f * (l[c] + r[c]);
  const float* __restrict__ base = x + (size_t)s * SLICE_F;
  const int lane = threadIdx.x & 31;
  const int wave = threadIdx.x >> 5;   // 8 waves per 256-thread block (wave32)

  v8i acc = {0, 0, 0, 0, 0, 0, 0, 0};
  const v8i bones = {0x01010101, 0x01010101, 0x01010101, 0x01010101,
                     0x01010101, 0x01010101, 0x01010101, 0x01010101};

  // full 1024-float chunks: waves 0-3 take {w, w+8}, waves 4-7 take {w}
  for (int chunk = wave; chunk < FULL_CHUNKS; chunk += 8) {
    const int cb = chunk * 1024;
    f32x4 f[8];
#pragma unroll
    for (int t = 0; t < 8; ++t)        // coalesced: 128 contiguous floats / inst
      f[t] = *(const f32x4*)(base + cb + t * 128 + lane * 4);
    v8i a;
#pragma unroll
    for (int t = 0; t < 8; ++t) a[t] = pack4(f[t], pivot);
    acc = __builtin_amdgcn_wmma_i32_16x16x64_iu8(false, a, false, bones, acc,
                                                 false, false);
  }

  // peeled tail: 256 floats -> only t=0,1 valid (all lanes in range), rest 0
  if (wave == 7) {                     // wave-uniform branch
    f32x4 f0 = *(const f32x4*)(base + TAIL_BASE + 0 * 128 + lane * 4);
    f32x4 f1 = *(const f32x4*)(base + TAIL_BASE + 1 * 128 + lane * 4);
    v8i a = {0, 0, 0, 0, 0, 0, 0, 0};
    a[0] = pack4(f0, pivot);
    a[1] = pack4(f1, pivot);
    acc = __builtin_amdgcn_wmma_i32_16x16x64_iu8(false, a, false, bones, acc,
                                                 false, false);
  }

  // D[m][n] identical across n. Lanes 0-15 hold M=vgpr, lanes 16-31 M=vgpr+8.
  int part = acc[0] + acc[1] + acc[2] + acc[3] + acc[4] + acc[5] + acc[6] + acc[7];
  const int wtotal = __builtin_amdgcn_readlane(part, 0) +
                     __builtin_amdgcn_readlane(part, 16);

  __shared__ int swsum[8];
  if (lane == 0) swsum[wave] = wtotal;
  __syncthreads();
  if (threadIdx.x == 0) {
    int t = 0;
#pragma unroll
    for (int w = 0; w < 8; ++w) t += swsum[w];
    atomicAdd(&cnt[c], (unsigned)t);
  }
}

// ---------------- bisection step: exact jax.lax.while_loop semantics ---------
__global__ void k_step(unsigned* flag, unsigned* cnt, float* l, float* r) {
  if (*flag) return;
  const int c = threadIdx.x;           // 128 threads
  float lv = l[c], rv = r[c];
  const float per = (float)cnt[c] / CH_SIZE;     // exact division, as reference
  const bool ok = ((per >= P_LO_F) && (per <= P_HI_F)) || ((rv - lv) <= 1.0f);

  __shared__ int s_all;
  if (c == 0) s_all = 1;
  __syncthreads();
  if (!ok) s_all = 0;                  // benign multi-write of 0
  __syncthreads();
  if (s_all) {                         // cond_fn false -> loop exits, freeze l/r
    if (c == 0) *flag = 1u;
    return;
  }
  const float pivot = 0.5f * (lv + rv);
  if (per > P_HI_F) lv = pivot;
  if (per < P_LO_F) rv = pivot;
  l[c] = lv; r[c] = rv; cnt[c] = 0u;   // reset for next counting pass
}

// ---------------- finalize per-channel scale ---------------------------------
__global__ void k_fin(const float* l, const float* r, float* sc) {
  const int c = threadIdx.x;           // 128 threads
  const float upper  = 0.5f * (l[c] + r[c]);
  const float rshift = rintf(8.0f - log2f(upper));   // round-half-even == jnp.round
  sc[c] = exp2f(rshift);
}

// ---------------- quantize: double-buffered async-to-LDS + NT stores ---------
__global__ void k_quant(const float* __restrict__ x, const float* __restrict__ sc,
                        float* __restrict__ out) {
  const int s = blockIdx.x;
  const int c = s & (C_CH - 1);
  const float scale = sc[c];
  const size_t base = (size_t)s * SLICE_F;
  const f32x4* __restrict__ src = (const f32x4*)(x + base);
  f32x4* __restrict__ dst = (f32x4*)(out + base);

  __shared__ f32x4 tile[2][256];
  unsigned ldsoff[2];
  ldsoff[0] = (unsigned)(uintptr_t)(&tile[0][threadIdx.x]);
  ldsoff[1] = (unsigned)(uintptr_t)(&tile[1][threadIdx.x]);

  int i = threadIdx.x;
  if (i < SLICE_V4) {                       // wave-uniform (3136 = 12.25*256)
    const f32x4* gp = src + i;
    asm volatile("global_load_async_to_lds_b128 %0, %1, off"
                 :: "v"(ldsoff[0]), "v"(gp) : "memory");
  }
  int buf = 0;
  for (; i < SLICE_V4; i += 256) {
    const int nxt = i + 256;
    const bool more = nxt < SLICE_V4;       // wave-uniform
    if (more) {                             // prefetch next tile into other buf
      const f32x4* gp = src + nxt;
      asm volatile("global_load_async_to_lds_b128 %0, %1, off"
                   :: "v"(ldsoff[buf ^ 1]), "v"(gp) : "memory");
      asm volatile("s_wait_asynccnt 0x1" ::: "memory");  // current tile landed
    } else {
      asm volatile("s_wait_asynccnt 0x0" ::: "memory");
    }
    f32x4 v = tile[buf][threadIdx.x];
    f32x4 q;
    q.x = fminf(fmaxf(floorf(v.x * scale), 0.0f), Q_MAX_F);
    q.y = fminf(fmaxf(floorf(v.y * scale), 0.0f), Q_MAX_F);
    q.z = fminf(fmaxf(floorf(v.z * scale), 0.0f), Q_MAX_F);
    q.w = fminf(fmaxf(floorf(v.w * scale), 0.0f), Q_MAX_F);
    // NT store: keep the L2-resident input from being evicted by the output.
    __builtin_nontemporal_store(q, dst + i);
    buf ^= 1;
  }
}

// ---------------- host-side launch sequence (graph-capture safe) -------------
extern "C" void kernel_launch(void* const* d_in, const int* in_sizes, int n_in,
                              void* d_out, int out_size, void* d_ws, size_t ws_size,
                              hipStream_t stream) {
  (void)in_sizes; (void)n_in; (void)out_size; (void)ws_size;
  const float* x = (const float*)d_in[0];
  float* out = (float*)d_out;

  unsigned* flag = (unsigned*)d_ws;        // [1]
  unsigned* cnt  = flag + 1;               // [128]
  unsigned* gmax = cnt + C_CH;             // [128] float bits
  float*    l    = (float*)(gmax + C_CH);  // [128]
  float*    r    = l + C_CH;               // [128]
  float*    sc   = r + C_CH;               // [128]

  k_init<<<1, 256, 0, stream>>>(flag, cnt, gmax);
  k_max <<<SLICES, 256, 0, stream>>>(x, gmax);
  k_seed<<<1, 128, 0, stream>>>(gmax, l, r, cnt, flag);

  // Fixed 16 iterations (actual convergence <= ~5; flagged passes no-op).
  for (int it = 0; it < 16; ++it) {
    k_count<<<SLICES, 256, 0, stream>>>(x, l, r, flag, cnt);
    k_step <<<1, 128, 0, stream>>>(flag, cnt, l, r);
  }

  k_fin  <<<1, 128, 0, stream>>>(l, r, sc);
  k_quant<<<SLICES, 256, 0, stream>>>(x, sc, out);
}
